// TreeMamba2_25907242729927
// MI455X (gfx1250) — matrix operand order repaired and checked by reading, not compiled, not run on hardware
//
#include <hip/hip_runtime.h>

// ---------------------------------------------------------------------------
// TreeMamba2 block for MI455X (gfx1250, wave32, WMMA).
// All matmuls (in-proj, C·B^T, G~·dtx, B^T·dtx, C·prev^T, out-proj) run on
// v_wmma_f32_16x16x32_bf16 with f32 accumulation. Scalar math (cumsum, exp,
// softplus, SiLU, RMSNorm) stays f32. bf16 stored as raw uint16 (RNE round).
// GEMM LDS fills use CDNA5 async global->LDS copies (ASYNCcnt) with LDS
// double buffering when the toolchain exposes the builtin.
// ---------------------------------------------------------------------------

typedef __bf16 bf16x16 __attribute__((ext_vector_type(16)));
typedef float  f32x8   __attribute__((ext_vector_type(8)));
typedef unsigned short u16x8 __attribute__((ext_vector_type(8)));
typedef int    v4i     __attribute__((vector_size(16)));   // matches builtin sig

#if defined(__AMDGCN__) && __has_builtin(__builtin_amdgcn_global_load_async_to_lds_b128)
#define HAVE_ASYNC_LDS 1
#else
#define HAVE_ASYNC_LDS 0
#endif

union Frag {              // one WMMA 16x16(x32) bf16 operand per lane
    bf16x16        v;
    u16x8          h[2];
    unsigned short s[16];
};

__device__ __forceinline__ unsigned short f2b(float f) {
    union { float f; unsigned u; } c; c.f = f;
    unsigned u = c.u;
    unsigned r = u + 0x7FFFu + ((u >> 16) & 1u);   // round-to-nearest-even
    return (unsigned short)(r >> 16);
}
__device__ __forceinline__ float b2f(unsigned short h) {
    union { unsigned u; float f; } c; c.u = ((unsigned)h) << 16;
    return c.f;
}
__device__ __forceinline__ f32x8 wmma_bf16(bf16x16 a, bf16x16 b, f32x8 c) {
    return __builtin_amdgcn_wmma_f32_16x16x32_bf16(
        /*neg_a=*/false, a, /*neg_b=*/false, b,
        /*c_mod=*/(short)0, c, /*reuse_a=*/false, /*reuse_b=*/false);
}

// 16-byte global -> LDS copy. Async (no VGPR round-trip, ASYNCcnt tracked)
// when available, else plain load+store.
__device__ __forceinline__ void cp_g2l_b128(const unsigned short* gsrc,
                                            unsigned short* ldst) {
#if HAVE_ASYNC_LDS
    unsigned short* gnc = (unsigned short*)gsrc;   // drop const in generic AS
    __builtin_amdgcn_global_load_async_to_lds_b128(
        (__attribute__((address_space(1))) v4i*)gnc,
        (__attribute__((address_space(3))) v4i*)ldst,
        /*offset=*/0, /*cpol=*/0);
#else
    *(u16x8*)ldst = *(const u16x8*)gsrc;
#endif
}
__device__ __forceinline__ void wait_g2l() {
#if HAVE_ASYNC_LDS
#if __has_builtin(__builtin_amdgcn_s_wait_asynccnt)
    __builtin_amdgcn_s_wait_asynccnt(0);
#else
    asm volatile("s_wait_asynccnt 0" ::: "memory");
#endif
#endif
}

// Model constants
#define SEQ     2048
#define BATCH_N 2
#define DMODEL  1024
#define DSSM    2048
#define NHEADS  32
#define HEADD   64
#define DSTATE  128
#define CONVD   2304
#define DPROJ   4384
#define CHUNKL  256
#define NCHUNK  8
#define ROWS    (BATCH_N * SEQ)          // 4096

// ---------------------------------------------------------------------------
// f32 -> bf16(u16) conversion, grid-strided
// ---------------------------------------------------------------------------
__global__ void cvt_bf16_kernel(const float* __restrict__ src,
                                unsigned short* __restrict__ dst, int n) {
    int i = blockIdx.x * blockDim.x + threadIdx.x;
    for (; i < n; i += gridDim.x * blockDim.x) dst[i] = f2b(src[i]);
}

// ---------------------------------------------------------------------------
// Generic NT GEMM:  C[M,N] f32 = A[M,K] bf16  *  B[N,K]^T bf16
// block = 256 threads (8 waves), tile 128x128, BK=32 (one WMMA k-step).
// Double-buffered LDS; fills via async global->LDS copies.
// ---------------------------------------------------------------------------
__global__ __launch_bounds__(256) void gemm_nt_bf16(
    const unsigned short* __restrict__ A,
    const unsigned short* __restrict__ B,
    float* __restrict__ C, int M, int N, int K)
{
    constexpr int BM = 128, BN = 128, BK = 32, LDT = 40;  // LDS row pad -> 80B
    __shared__ __attribute__((aligned(16))) unsigned short As[2][BM * LDT];
    __shared__ __attribute__((aligned(16))) unsigned short Bs[2][BN * LDT];

    const int tid = threadIdx.x, lane = tid & 31, w = tid >> 5;
    const int g = lane >> 4, ln = lane & 15, kb = g * 8;
    const int bm = blockIdx.y * BM, bn = blockIdx.x * BN;
    const int wm = (w >> 2) * 64;   // 2 wave-rows of 64
    const int wn = (w & 3) * 32;    // 4 wave-cols of 32
    const bool fullB = (bn + BN <= N);   // remainder tiles need zero-fill

    // each thread owns two 8-element (16B) fill slots per matrix:
    const int fr = tid >> 2;              // rows fr and fr+64
    const int fc = (tid & 3) * 8;         // col group within BK

    f32x8 acc[4][2] = {};
    const int nk = K / BK;

    // ---- fill helpers (inlined twice: prologue + steady state) ----
    #define FILL_TILE(k0, bf)                                                  \
    do {                                                                       \
        cp_g2l_b128(&A[(size_t)(bm + fr) * K + (k0) + fc],                     \
                    &As[bf][fr * LDT + fc]);                                   \
        cp_g2l_b128(&A[(size_t)(bm + fr + 64) * K + (k0) + fc],                \
                    &As[bf][(fr + 64) * LDT + fc]);                            \
        if (fullB) {                                                           \
            cp_g2l_b128(&B[(size_t)(bn + fr) * K + (k0) + fc],                 \
                        &Bs[bf][fr * LDT + fc]);                               \
            cp_g2l_b128(&B[(size_t)(bn + fr + 64) * K + (k0) + fc],            \
                        &Bs[bf][(fr + 64) * LDT + fc]);                        \
        } else {                                                               \
            u16x8 bv0 = {}, bv1 = {};                                          \
            if (bn + fr < N)                                                   \
                bv0 = *(const u16x8*)(&B[(size_t)(bn + fr) * K + (k0) + fc]);  \
            if (bn + fr + 64 < N)                                              \
                bv1 = *(const u16x8*)(&B[(size_t)(bn + fr + 64) * K + (k0) + fc]); \
            *(u16x8*)(&Bs[bf][fr * LDT + fc]) = bv0;                           \
            *(u16x8*)(&Bs[bf][(fr + 64) * LDT + fc]) = bv1;                    \
        }                                                                      \
    } while (0)

    FILL_TILE(0, 0);
    wait_g2l();
    __syncthreads();

    for (int i = 0; i < nk; ++i) {
        const int cur = i & 1;
        if (i + 1 < nk) FILL_TILE((i + 1) * BK, cur ^ 1);   // async into nxt
        if (i + 2 < nk)   // CDNA5 global_prefetch of the tile after that
            __builtin_prefetch(&A[(size_t)(bm + fr) * K + (i + 2) * BK], 0, 1);

        Frag fa[4], fb[2];
        #pragma unroll
        for (int mt = 0; mt < 4; ++mt) {
            const unsigned short* rp = &As[cur][(wm + mt * 16 + ln) * LDT];
            fa[mt].h[0] = *(const u16x8*)(rp + g * 16);
            fa[mt].h[1] = *(const u16x8*)(rp + g * 16 + 8);
        }
        #pragma unroll
        for (int nt = 0; nt < 2; ++nt) {
            const unsigned short* rp = &Bs[cur][(wn + nt * 16 + ln) * LDT];
            fb[nt].h[0] = *(const u16x8*)(rp + g * 16);
            fb[nt].h[1] = *(const u16x8*)(rp + g * 16 + 8);
        }
        #pragma unroll
        for (int mt = 0; mt < 4; ++mt)
            #pragma unroll
            for (int nt = 0; nt < 2; ++nt)
                acc[mt][nt] = wmma_bf16(fa[mt].v, fb[nt].v, acc[mt][nt]);

        wait_g2l();        // our async fills of nxt are done
        __syncthreads();   // everyone finished reading cur / writing nxt
    }
    #undef FILL_TILE

    // --- store (C/D layout: lane = col, vgpr r = row 8g+r) ---
    #pragma unroll
    for (int mt = 0; mt < 4; ++mt)
        #pragma unroll
        for (int nt = 0; nt < 2; ++nt) {
            int col = bn + wn + nt * 16 + ln;
            if (col >= N) continue;
            #pragma unroll
            for (int r = 0; r < 8; ++r) {
                int row = bm + wm + mt * 16 + 8 * g + r;
                C[(size_t)row * N + col] = acc[mt][nt][r];
            }
        }
}

// ---------------------------------------------------------------------------
// dt = softplus(dt_raw + dt_bias),  one thread per (b,l,h)
// ---------------------------------------------------------------------------
__global__ void dt_kernel(const float* __restrict__ zxbcdt,
                          const float* __restrict__ dt_bias,
                          float* __restrict__ dt_ws) {
    int i = blockIdx.x * blockDim.x + threadIdx.x;     // < ROWS*NHEADS
    int h = i & 31;
    size_t row = (size_t)(i >> 5);
    float x = zxbcdt[row * DPROJ + (DSSM + CONVD) + h] + dt_bias[h];
    dt_ws[i] = (x > 20.f) ? x : log1pf(expf(x));
}

// ---------------------------------------------------------------------------
// Tree conv (gather-4, depthwise k=4) + bias + SiLU, then route to
// x (bf16), dtx^T (bf16, (b,h,p,l)), B row-major + transposed, C row-major.
// grid = (9, SEQ, BATCH), block = 256 (ch = bx*256+tid, 9*256 = CONVD)
// ---------------------------------------------------------------------------
__global__ __launch_bounds__(256) void conv_kernel(
    const float* __restrict__ zxbcdt, const int* __restrict__ conv_indices,
    const float* __restrict__ conv_w, const float* __restrict__ conv_b,
    const float* __restrict__ dt_ws,
    unsigned short* __restrict__ xb, unsigned short* __restrict__ dtxT,
    unsigned short* __restrict__ Bb, unsigned short* __restrict__ Bt,
    unsigned short* __restrict__ Cb)
{
    const int ch = blockIdx.x * 256 + threadIdx.x;
    const int l = blockIdx.y, b = blockIdx.z;
    const int* ip = conv_indices + (size_t)b * (4 * SEQ) + 4 * l;
    float acc = conv_b[ch];
    #pragma unroll
    for (int k = 0; k < 4; ++k) {
        int j = ip[k];                    // 0 selects the zero pad
        float v = 0.f;
        if (j > 0)
            v = zxbcdt[((size_t)b * SEQ + (j - 1)) * DPROJ + DSSM + ch];
        acc += v * conv_w[ch * 4 + k];
    }
    float s = acc / (1.f + expf(-acc));   // SiLU
    size_t row = (size_t)b * SEQ + l;
    if (ch < DSSM) {
        int h = ch >> 6, p = ch & 63;
        xb[row * DSSM + ch] = f2b(s);
        float dtv = dt_ws[row * NHEADS + h];
        dtxT[(((size_t)b * NHEADS + h) * HEADD + p) * SEQ + l] = f2b(s * dtv);
    } else if (ch < DSSM + DSTATE) {
        int n = ch - DSSM;
        unsigned short hv = f2b(s);
        Bb[row * DSTATE + n] = hv;
        Bt[((size_t)b * DSTATE + n) * SEQ + l] = hv;
    } else {
        Cb[row * DSTATE + (ch - DSSM - DSTATE)] = f2b(s);
    }
}

// ---------------------------------------------------------------------------
// Per-(b,chunk,head): cumsum(dt*A) -> Acum; states = B^T (decay o dtx)
// via WMMA:  ST[n,p] (M=128, N=64, K=256).  grid = (NHEADS, NCHUNK, BATCH)
// ---------------------------------------------------------------------------
__global__ __launch_bounds__(256) void ssd_states_kernel(
    const float* __restrict__ dt_ws, const float* __restrict__ A_log,
    const unsigned short* __restrict__ Bt, const unsigned short* __restrict__ dtxT,
    float* __restrict__ Acum_ws, float* __restrict__ states_ws)
{
    const int h = blockIdx.x, c = blockIdx.y, b = blockIdx.z;
    const int tid = threadIdx.x, lane = tid & 31, w = tid >> 5;
    const int g = lane >> 4, ln = lane & 15, kb = g * 8;
    __shared__ float sA[CHUNKL];
    __shared__ float sDec[CHUNKL];

    const float A_h = -expf(A_log[h]);
    float v = dt_ws[((size_t)b * SEQ + c * CHUNKL + tid) * NHEADS + h] * A_h;
    sA[tid] = v;
    __syncthreads();
    // inclusive Hillis-Steele scan, 256 elements
    for (int off = 1; off < CHUNKL; off <<= 1) {
        float t = (tid >= off) ? sA[tid - off] : 0.f;
        __syncthreads();
        sA[tid] += t;
        __syncthreads();
    }
    float last = sA[CHUNKL - 1];
    sDec[tid] = expf(last - sA[tid]);                 // decay_states (<=1)
    Acum_ws[(((size_t)b * NCHUNK + c) * NHEADS + h) * CHUNKL + tid] = sA[tid];
    __syncthreads();

    // ST[n,p] = sum_s Bt[n,s] * (dtxT[p,s] * decay[s]);  wave w -> n-tile w
    const unsigned short* arow =
        Bt + ((size_t)b * DSTATE + (w * 16 + ln)) * SEQ + c * CHUNKL;
    f32x8 acc[4] = {};
    for (int k0 = 0; k0 < CHUNKL; k0 += 32) {
        Frag fa;
        fa.h[0] = *(const u16x8*)(arow + k0 + g * 16);
        fa.h[1] = *(const u16x8*)(arow + k0 + g * 16 + 8);
        #pragma unroll
        for (int nt = 0; nt < 4; ++nt) {
            int p = nt * 16 + ln;
            const unsigned short* rp =
                dtxT + (((size_t)b * NHEADS + h) * HEADD + p) * SEQ +
                c * CHUNKL + k0 + g * 16;
            Frag t2; t2.h[0] = *(const u16x8*)rp; t2.h[1] = *(const u16x8*)(rp + 8);
            Frag fb;
            #pragma unroll
            for (int e = 0; e < 16; ++e)
                fb.s[e] = f2b(b2f(t2.s[e]) * sDec[k0 + g * 16 + e]);
            acc[nt] = wmma_bf16(fa.v, fb.v, acc[nt]);
        }
    }
    size_t base = (((size_t)b * NCHUNK + c) * NHEADS + h) * (DSTATE * HEADD);
    #pragma unroll
    for (int nt = 0; nt < 4; ++nt)
        #pragma unroll
        for (int r = 0; r < 8; ++r)
            states_ws[base + (size_t)(w * 16 + 8 * g + r) * HEADD + nt * 16 + ln]
                = acc[nt][r];
}

// ---------------------------------------------------------------------------
// Sequential inter-chunk scan over NCHUNK chunks; emits prev-state entering
// each chunk, transposed to (p,n) in bf16.  grid = (NHEADS, BATCH)
// ---------------------------------------------------------------------------
__global__ __launch_bounds__(256) void scan_kernel(
    const float* __restrict__ states_ws, const float* __restrict__ Acum_ws,
    unsigned short* __restrict__ prevT)
{
    const int h = blockIdx.x, b = blockIdx.y, tid = threadIdx.x;
    float run[32];
    #pragma unroll
    for (int j = 0; j < 32; ++j) run[j] = 0.f;
    for (int c = 0; c < NCHUNK; ++c) {
        size_t hb = (((size_t)b * NCHUNK + c) * NHEADS + h);
        float cd = expf(Acum_ws[hb * CHUNKL + (CHUNKL - 1)]);   // chunk decay
        size_t base = hb * (DSTATE * HEADD);
        #pragma unroll
        for (int j = 0; j < 32; ++j) {
            int idx = j * 256 + tid;          // idx = n*64 + p
            int n = idx >> 6, p = idx & 63;
            prevT[base + (size_t)p * DSTATE + n] = f2b(run[j]);
            run[j] = run[j] * cd + states_ws[base + idx];
        }
    }
}

// ---------------------------------------------------------------------------
// Per-(b,chunk,head) output:  G = C B^T (WMMA), masked exp(Acum[t]-Acum[s]),
// Yd = G~ dtx (WMMA), Yoff = (exp(Acum[t]) C) prev^T (WMMA), + D*x.
// Processed in 4 passes of 64 rows; G~ staged bf16 in LDS.
// grid = (NHEADS, NCHUNK, BATCH)
// ---------------------------------------------------------------------------
__global__ __launch_bounds__(256) void ssd_out_kernel(
    const unsigned short* __restrict__ Cb, const unsigned short* __restrict__ Bb,
    const unsigned short* __restrict__ dtxT, const unsigned short* __restrict__ prevT,
    const unsigned short* __restrict__ xb, const float* __restrict__ Acum_ws,
    const float* __restrict__ Dvec, float* __restrict__ y)
{
    const int h = blockIdx.x, c = blockIdx.y, b = blockIdx.z;
    const int tid = threadIdx.x, lane = tid & 31, w = tid >> 5;
    const int g = lane >> 4, ln = lane & 15, kb = g * 8;
    constexpr int GLD = 264;                        // 256 + 8 pad (u16)
    __shared__ float sAc[CHUNKL];
    __shared__ float sEx[CHUNKL];
    __shared__ __attribute__((aligned(16))) unsigned short Gs[64 * GLD];

    size_t hb = (((size_t)b * NCHUNK + c) * NHEADS + h);
    sAc[tid] = Acum_ws[hb * CHUNKL + tid];
    sEx[tid] = expf(sAc[tid]);
    const float D_h = Dvec[h];
    const size_t lbase = (size_t)b * SEQ + c * CHUNKL;
    __syncthreads();

    for (int pass = 0; pass < 4; ++pass) {
        const int t0 = pass * 64;
        // ---- G = C B^T for rows [t0, t0+64) ----
        const int mtg = w & 3;              // 4 row-tiles of 16
        const int sgrp = (w >> 2) * 8;      // 8 col(s)-tiles each
        const int trow = t0 + mtg * 16 + ln;
        const unsigned short* crow = Cb + (lbase + trow) * DSTATE;
        f32x8 accG[8] = {};
        for (int k0 = 0; k0 < DSTATE; k0 += 32) {
            Frag fa;
            fa.h[0] = *(const u16x8*)(crow + k0 + g * 16);
            fa.h[1] = *(const u16x8*)(crow + k0 + g * 16 + 8);
            #pragma unroll
            for (int j = 0; j < 8; ++j) {
                int srow = (sgrp + j) * 16 + ln;
                const unsigned short* brow =
                    Bb + (lbase + srow) * DSTATE + k0 + g * 16;
                Frag fb; fb.h[0] = *(const u16x8*)brow;
                fb.h[1] = *(const u16x8*)(brow + 8);
                accG[j] = wmma_bf16(fa.v, fb.v, accG[j]);
            }
        }
        // causal mask * exp(Acum[t]-Acum[s]) -> LDS bf16
        #pragma unroll
        for (int j = 0; j < 8; ++j) {
            int s = (sgrp + j) * 16 + ln;
            #pragma unroll
            for (int r = 0; r < 8; ++r) {
                int tl = mtg * 16 + 8 * g + r;
                int t = t0 + tl;
                float val = (s <= t) ? accG[j][r] * expf(sAc[t] - sAc[s]) : 0.f;
                Gs[tl * GLD + s] = f2b(val);
            }
        }
        __syncthreads();

        // ---- Yd + Yoff for rows [t0, t0+64) ----
        const int mt2 = w >> 1;             // 4 row-tiles of 16
        const int np0 = (w & 1) * 2;        // 2 p-tiles each
        f32x8 accY[2] = {};
        // Yd: K = 256 over s, A from LDS G~, B from dtx^T rows
        const unsigned short* grow = &Gs[(mt2 * 16 + ln) * GLD];
        for (int k0 = 0; k0 < CHUNKL; k0 += 32) {
            Frag fa;
            fa.h[0] = *(const u16x8*)(grow + k0 + g * 16);
            fa.h[1] = *(const u16x8*)(grow + k0 + g * 16 + 8);
            #pragma unroll
            for (int nt = 0; nt < 2; ++nt) {
                int p = (np0 + nt) * 16 + ln;
                const unsigned short* rp =
                    dtxT + (((size_t)b * NHEADS + h) * HEADD + p) * SEQ +
                    c * CHUNKL + k0 + g * 16;
                Frag fb; fb.h[0] = *(const u16x8*)rp;
                fb.h[1] = *(const u16x8*)(rp + 8);
                accY[nt] = wmma_bf16(fa.v, fb.v, accY[nt]);
            }
        }
        // Yoff: K = 128 over n, A = exp(Acum[t]) * C row, B from prev^T rows
        const int t2 = t0 + mt2 * 16 + ln;
        const unsigned short* crow2 = Cb + (lbase + t2) * DSTATE;
        const float sc = sEx[t2];
        for (int k0 = 0; k0 < DSTATE; k0 += 32) {
            Frag fa;
            fa.h[0] = *(const u16x8*)(crow2 + k0 + g * 16);
            fa.h[1] = *(const u16x8*)(crow2 + k0 + g * 16 + 8);
            #pragma unroll
            for (int e = 0; e < 16; ++e)
                fa.s[e] = f2b(b2f(fa.s[e]) * sc);
            #pragma unroll
            for (int nt = 0; nt < 2; ++nt) {
                int p = (np0 + nt) * 16 + ln;
                const unsigned short* rp =
                    prevT + hb * (DSTATE * HEADD) + (size_t)p * DSTATE +
                    k0 + g * 16;
                Frag fb; fb.h[0] = *(const u16x8*)rp;
                fb.h[1] = *(const u16x8*)(rp + 8);
                accY[nt] = wmma_bf16(fa.v, fb.v, accY[nt]);
            }
        }
        // combine + D*x, write y (b,l,h,p) f32
        #pragma unroll
        for (int nt = 0; nt < 2; ++nt) {
            int p = (np0 + nt) * 16 + ln;
            #pragma unroll
            for (int r = 0; r < 8; ++r) {
                int t = t0 + mt2 * 16 + 8 * g + r;
                size_t oi = ((lbase + t) * NHEADS + h) * HEADD + p;
                y[oi] = accY[nt][r] + D_h * b2f(xb[oi]);
            }
        }
        __syncthreads();   // Gs reused next pass
    }
}

// ---------------------------------------------------------------------------
// Gated RMSNorm per (b,l) row of 2048: yz = y*silu(z); y' = yz*rsqrt(mean(yz^2)
// + eps)*norm_w  -> bf16 for the out-proj GEMM.  grid = ROWS, block = 256
// ---------------------------------------------------------------------------
__global__ __launch_bounds__(256) void norm_kernel(
    const float* __restrict__ y, const float* __restrict__ zxbcdt,
    const float* __restrict__ norm_w, unsigned short* __restrict__ ynorm)
{
    __shared__ float red[256];
    const size_t row = blockIdx.x;
    const int tid = threadIdx.x;
    float t[8], ss = 0.f;
    #pragma unroll
    for (int j = 0; j < 8; ++j) {
        int ch = tid + j * 256;
        float z = zxbcdt[row * DPROJ + ch];            // z = first 2048 cols
        float v = y[row * DSSM + ch] * (z / (1.f + expf(-z)));
        t[j] = v; ss += v * v;
    }
    red[tid] = ss; __syncthreads();
    for (int off = 128; off > 0; off >>= 1) {
        if (tid < off) red[tid] += red[tid + off];
        __syncthreads();
    }
    float scale = rsqrtf(red[0] / (float)DSSM + 1e-5f);
    #pragma unroll
    for (int j = 0; j < 8; ++j) {
        int ch = tid + j * 256;
        ynorm[row * DSSM + ch] = f2b(t[j] * scale * norm_w[ch]);
    }
}

// ---------------------------------------------------------------------------
// Host launcher
// ---------------------------------------------------------------------------
extern "C" void kernel_launch(void* const* d_in, const int* in_sizes, int n_in,
                              void* d_out, int out_size, void* d_ws, size_t ws_size,
                              hipStream_t stream) {
    (void)in_sizes; (void)n_in; (void)out_size; (void)ws_size;
    const float* u            = (const float*)d_in[0];
    const int*   conv_indices = (const int*)  d_in[1];
    const float* W_in         = (const float*)d_in[2];
    const float* conv_w       = (const float*)d_in[3];
    const float* conv_b       = (const float*)d_in[4];
    const float* dt_bias      = (const float*)d_in[5];
    const float* A_log        = (const float*)d_in[6];
    const float* Dv           = (const float*)d_in[7];
    const float* norm_w       = (const float*)d_in[8];
    const float* W_out        = (const float*)d_in[9];
    float* out = (float*)d_out;

    // workspace carve-up (each 256B aligned); total ~207 MB
    char* ws = (char*)d_ws;
    size_t off = 0;
    auto carve = [&](size_t bytes) -> char* {
        char* p = ws + off;
        off += (bytes + 255) & ~(size_t)255;
        return p;
    };
    unsigned short* u16u    = (unsigned short*)carve((size_t)ROWS * DMODEL * 2);
    unsigned short* win16   = (unsigned short*)carve((size_t)DPROJ * DMODEL * 2);
    unsigned short* wout16  = (unsigned short*)carve((size_t)DMODEL * DSSM * 2);
    float*          zxbcdt  = (float*)         carve((size_t)ROWS * DPROJ * 4);
    float*          dt_ws   = (float*)         carve((size_t)ROWS * NHEADS * 4);
    unsigned short* xb      = (unsigned short*)carve((size_t)ROWS * DSSM * 2);
    unsigned short* dtxT    = (unsigned short*)carve((size_t)ROWS * DSSM * 2);
    unsigned short* Bb      = (unsigned short*)carve((size_t)ROWS * DSTATE * 2);
    unsigned short* Bt      = (unsigned short*)carve((size_t)ROWS * DSTATE * 2);
    unsigned short* Cb      = (unsigned short*)carve((size_t)ROWS * DSTATE * 2);
    float*          Acum_ws = (float*)         carve((size_t)BATCH_N * NCHUNK * NHEADS * CHUNKL * 4);
    float*          states  = (float*)         carve((size_t)BATCH_N * NCHUNK * NHEADS * DSTATE * HEADD * 4);
    unsigned short* prevT   = (unsigned short*)carve((size_t)BATCH_N * NCHUNK * NHEADS * DSTATE * HEADD * 2);
    float*          y_ws    = (float*)         carve((size_t)ROWS * DSSM * 4);
    unsigned short* ynorm   = (unsigned short*)carve((size_t)ROWS * DSSM * 2);

    // 0) f32 -> bf16 conversions
    cvt_bf16_kernel<<<2048, 256, 0, stream>>>(u, u16u, ROWS * DMODEL);
    cvt_bf16_kernel<<<2048, 256, 0, stream>>>(W_in, win16, DPROJ * DMODEL);
    cvt_bf16_kernel<<<2048, 256, 0, stream>>>(W_out, wout16, DMODEL * DSSM);

    // 1) in-projection: zxbcdt = u @ W_in^T   (M=4096, N=4384, K=1024)
    gemm_nt_bf16<<<dim3((DPROJ + 127) / 128, ROWS / 128), 256, 0, stream>>>(
        u16u, win16, zxbcdt, ROWS, DPROJ, DMODEL);

    // 2) dt = softplus(dt_raw + dt_bias)
    dt_kernel<<<(ROWS * NHEADS) / 256, 256, 0, stream>>>(zxbcdt, dt_bias, dt_ws);

    // 3) tree conv + SiLU + operand layout
    conv_kernel<<<dim3(CONVD / 256, SEQ, BATCH_N), 256, 0, stream>>>(
        zxbcdt, conv_indices, conv_w, conv_b, dt_ws, xb, dtxT, Bb, Bt, Cb);

    // 4) per-chunk cumsum + states (WMMA)
    ssd_states_kernel<<<dim3(NHEADS, NCHUNK, BATCH_N), 256, 0, stream>>>(
        dt_ws, A_log, Bt, dtxT, Acum_ws, states);

    // 5) inter-chunk scan -> prev^T
    scan_kernel<<<dim3(NHEADS, BATCH_N), 256, 0, stream>>>(states, Acum_ws, prevT);

    // 6) intra + inter chunk outputs (WMMA) -> y
    ssd_out_kernel<<<dim3(NHEADS, NCHUNK, BATCH_N), 256, 0, stream>>>(
        Cb, Bb, dtxT, prevT, xb, Acum_ws, Dv, y_ws);

    // 7) gated RMSNorm -> bf16
    norm_kernel<<<ROWS, 256, 0, stream>>>(y_ws, zxbcdt, norm_w, ynorm);

    // 8) out-projection: out = ynorm @ W_out^T  (M=4096, N=1024, K=2048)
    gemm_nt_bf16<<<dim3(DMODEL / 128, ROWS / 128), 256, 0, stream>>>(
        ynorm, wout16, out, ROWS, DMODEL, DSSM);
}
